// GraphIsomorphismNetwork_31009663877671
// MI455X (gfx1250) — matrix operand order repaired and checked by reading, not compiled, not run on hardware
//
#include <hip/hip_runtime.h>
#include <hip/hip_bf16.h>

// ---------------------------------------------------------------------------
// GIN forward on gfx1250 (MI455X).
//   * GEMMs: bf16 WMMA (f32 accumulate), operands pre-converted ONCE to bf16
//     (A row-major, B transposed K-major) so each k-step is 4x b128 loads +
//     one v_wmma_f32_16x16x32_bf16. K templated -> fully unrolled WMMA chain.
//   * Aggregation: edge-parallel global_atomic_add_f32; node features
//     (<= 51 MB) are resident in the 192 MB L2, so atomics resolve in L2.
// ---------------------------------------------------------------------------

typedef __attribute__((ext_vector_type(16))) __bf16       v16bf;
typedef __attribute__((ext_vector_type(8)))  float        v8f;
typedef __attribute__((ext_vector_type(8)))  unsigned int v8u;
typedef __attribute__((ext_vector_type(4)))  unsigned int v4u;

#define NNODES 100000
#define NEDGES 1600000
#define INF    128
#define HID    64
#define OUTF   32
#define NGRAPH 2048
#define BN_EPS 1e-5f

// ---- fp32 -> bf16 round-to-nearest-even ----------------------------------
__device__ __forceinline__ unsigned int bf16_bits(float f) {
  unsigned int u = __float_as_uint(f);
  return (u + 0x7FFFu + ((u >> 16) & 1u)) >> 16;
}
__device__ __forceinline__ unsigned int pack2bf(float a, float b) {
  return bf16_bits(a) | (bf16_bits(b) << 16);
}

// ---- packed f32 -> bf16 conversion (one pass per activation buffer) -------
__global__ void cvt_bf16_pk(const float2* __restrict__ src,
                            unsigned int* __restrict__ dst, size_t n2) {
  size_t t = (size_t)blockIdx.x * blockDim.x + threadIdx.x;
  size_t stride = (size_t)gridDim.x * blockDim.x;
  for (; t < n2; t += stride) {
    float2 v = src[t];
    dst[t] = pack2bf(v.x, v.y);
  }
}

// ---- weight convert + transpose: w[K][N] f32 -> wT[N][K] bf16 -------------
__global__ void wconv_t(const float* __restrict__ w, unsigned short* __restrict__ wt,
                        int K, int N) {
  int t = blockIdx.x * blockDim.x + threadIdx.x;
  if (t < K * N) {
    int k = t / N, n = t - k * N;
    wt[n * K + k] = (unsigned short)bf16_bits(w[t]);
  }
}

// ---------------------------------------------------------------------------
// GEMM: C[M,N] = act(A[M,K] @ B[K,N] + bias[N])
//   A  : bf16 row-major [M][K]
//   Bt : bf16 transposed [N][K]
//   Out: f32 (Cf) or bf16 (Cb) selected by which pointer is non-null.
// One wave32 per 16x16 tile; K is a template constant -> unrolled WMMA chain.
// ---------------------------------------------------------------------------
template <int K>
__global__ __launch_bounds__(256)
void gemm_wmma(const unsigned short* __restrict__ A,
               const unsigned short* __restrict__ Bt,
               const float* __restrict__ bias,
               float* __restrict__ Cf, unsigned short* __restrict__ Cb,
               int M, int N, int relu) {
  const int wave = threadIdx.x >> 5;
  const int lane = threadIdx.x & 31;
  const int nct  = N >> 4;
  const int tile = blockIdx.x * 8 + wave;
  const int tM   = tile / nct;
  const int tN   = tile - tM * nct;
  if (tM * 16 >= M) return;                 // wave-uniform: EXEC stays all-ones

  const int half = lane >> 4;
  const int l16  = lane & 15;
  const int m    = tM * 16 + l16;           // A row held by this lane
  const int n    = tN * 16 + l16;           // B column / D column of this lane

  const unsigned short* __restrict__ arow = A  + (size_t)m * K;
  const unsigned short* __restrict__ bcol = Bt + (size_t)n * K;

  v8f acc = {0.f, 0.f, 0.f, 0.f, 0.f, 0.f, 0.f, 0.f};

#pragma unroll
  for (int k0 = 0; k0 < K; k0 += 32) {
    // A 16x32 bf16 fragment (ISA 7.12.2): VGPR0-3 = K(half*8 .. +7),
    //                                     VGPR4-7 = K(16+half*8 .. +7)
    v4u a0 = *(const v4u*)(arow + k0 + half * 8);
    v4u a1 = *(const v4u*)(arow + k0 + 16 + half * 8);
    // B 32x16 bf16 fragment: VGPR0-3 = K(half*8 .. +7), VGPR4-7 = K(16+half*8 .. +7) at fixed column n
    v4u b0 = *(const v4u*)(bcol + k0 + half * 8);
    v4u b1 = *(const v4u*)(bcol + k0 + 16 + half * 8);
    v8u ar = {a0[0], a0[1], a0[2], a0[3], a1[0], a1[1], a1[2], a1[3]};
    v8u br = {b0[0], b0[1], b0[2], b0[3], b1[0], b1[1], b1[2], b1[3]};
    acc = __builtin_amdgcn_wmma_f32_16x16x32_bf16(
        false, __builtin_bit_cast(v16bf, ar),
        false, __builtin_bit_cast(v16bf, br),
        (short)0, acc, false, false);
  }

  const float bv = bias[n];
#pragma unroll
  for (int v = 0; v < 8; ++v) {
    int row = tM * 16 + v + 8 * half;       // D layout: row = vgpr + 8*half
    float val = acc[v] + bv;
    if (relu) val = fmaxf(val, 0.0f);
    if (Cb) Cb[(size_t)row * N + n] = (unsigned short)bf16_bits(val);
    else    Cf[(size_t)row * N + n] = val;
  }
}

// ---- elementwise float4 copy / zero ---------------------------------------
__global__ void copy_f4(const float4* __restrict__ src, float4* __restrict__ dst,
                        size_t n4) {
  size_t t = (size_t)blockIdx.x * blockDim.x + threadIdx.x;
  size_t stride = (size_t)gridDim.x * blockDim.x;
  for (; t < n4; t += stride) dst[t] = src[t];
}

__global__ void zero_f4(float4* __restrict__ dst, size_t n4) {
  size_t t = (size_t)blockIdx.x * blockDim.x + threadIdx.x;
  size_t stride = (size_t)gridDim.x * blockDim.x;
  float4 z = {0.f, 0.f, 0.f, 0.f};
  for (; t < n4; t += stride) dst[t] = z;
}

// ---- edge-parallel scatter-add: agg[dst] += h[src] (L2-resident atomics) --
__global__ void scatter_add_edges(const float* __restrict__ h,
                                  const int* __restrict__ eidx,
                                  float* __restrict__ agg, int E, int F) {
  const int* srci = eidx;         // edge_index[0]
  const int* dsti = eidx + E;     // edge_index[1]
  const int chunks = F >> 2;
  size_t total = (size_t)E * chunks;
  size_t t = (size_t)blockIdx.x * blockDim.x + threadIdx.x;
  size_t stride = (size_t)gridDim.x * blockDim.x;
  for (; t < total; t += stride) {
    size_t e = t / chunks;
    int c = (int)(t - e * chunks);
    int s = srci[e], d = dsti[e];
    float4 v = *(const float4*)(h + (size_t)s * F + c * 4);
    float* out = agg + (size_t)d * F + c * 4;
    atomicAdd(out + 0, v.x);
    atomicAdd(out + 1, v.y);
    atomicAdd(out + 2, v.z);
    atomicAdd(out + 3, v.w);
  }
}

// ---- BatchNorm: per-column mean / rstd over node rows ---------------------
__global__ __launch_bounds__(256)
void bn_stats(const float* __restrict__ h, int rows, int F,
              float* __restrict__ mean, float* __restrict__ rstd) {
  const int f = blockIdx.x;
  float s = 0.f, sq = 0.f;
  for (int i = threadIdx.x; i < rows; i += blockDim.x) {
    float v = h[(size_t)i * F + f];
    s += v;
    sq += v * v;
  }
  __shared__ float ss[256], sq2[256];
  ss[threadIdx.x] = s;
  sq2[threadIdx.x] = sq;
  __syncthreads();
  for (int off = 128; off > 0; off >>= 1) {
    if (threadIdx.x < off) {
      ss[threadIdx.x] += ss[threadIdx.x + off];
      sq2[threadIdx.x] += sq2[threadIdx.x + off];
    }
    __syncthreads();
  }
  if (threadIdx.x == 0) {
    float m = ss[0] / (float)rows;
    float var = sq2[0] / (float)rows - m * m;   // population variance (ddof=0)
    mean[f] = m;
    rstd[f] = rsqrtf(var + BN_EPS);
  }
}

__global__ void bn_apply(const float* __restrict__ h, float* __restrict__ out,
                         const float* __restrict__ mean, const float* __restrict__ rstd,
                         const float* __restrict__ gamma, const float* __restrict__ beta,
                         size_t total, int F) {
  size_t t = (size_t)blockIdx.x * blockDim.x + threadIdx.x;
  size_t stride = (size_t)gridDim.x * blockDim.x;
  for (; t < total; t += stride) {
    int f = (int)(t % F);
    out[t] = gamma[f] * (h[t] - mean[f]) * rstd[f] + beta[f];
  }
}

// ---- global_add_pool: g[batch[i]] += h[i] ---------------------------------
__global__ void pool_add(const float* __restrict__ h, const int* __restrict__ batch,
                         float* __restrict__ g, int rows, int F) {
  size_t total = (size_t)rows * F;
  size_t t = (size_t)blockIdx.x * blockDim.x + threadIdx.x;
  size_t stride = (size_t)gridDim.x * blockDim.x;
  for (; t < total; t += stride) {
    size_t i = t / F;
    int f = (int)(t - i * F);
    atomicAdd(&g[(size_t)batch[i] * F + f], h[t]);
  }
}

// ---------------------------------------------------------------------------
// Host-side orchestration
// ---------------------------------------------------------------------------
static inline int gemm_blocks(int M, int N) { return ((M / 16) * (N / 16) + 7) / 8; }

static inline void cvt(const float* src, unsigned short* dst, size_t n,
                       hipStream_t stream) {
  size_t n2 = n / 2;
  int blocks = (int)((n2 + 255) / 256);
  cvt_bf16_pk<<<blocks, 256, 0, stream>>>((const float2*)src, (unsigned int*)dst, n2);
}

extern "C" void kernel_launch(void* const* d_in, const int* in_sizes, int n_in,
                              void* d_out, int out_size, void* d_ws, size_t ws_size,
                              hipStream_t stream) {
  (void)in_sizes; (void)n_in; (void)out_size; (void)ws_size;

  const float* x        = (const float*)d_in[0];   // [100000,128]
  const int*   eidx     = (const int*)  d_in[1];   // [2,1600000]
  const int*   batch    = (const int*)  d_in[2];   // [100000]
  const float* c1w1     = (const float*)d_in[3];   // [128,64]
  const float* c1b1     = (const float*)d_in[4];
  const float* c1w2     = (const float*)d_in[5];   // [64,64]
  const float* c1b2     = (const float*)d_in[6];
  const float* cw1      = (const float*)d_in[7];   // [4,64,64]
  const float* cb1      = (const float*)d_in[8];
  const float* cw2      = (const float*)d_in[9];   // [4,64,64]
  const float* cb2      = (const float*)d_in[10];
  const float* bn_gamma = (const float*)d_in[11];  // [5,64]
  const float* bn_beta  = (const float*)d_in[12];
  const float* fc1_w    = (const float*)d_in[13];  // [64,64]
  const float* fc1_b    = (const float*)d_in[14];
  const float* fc2_w    = (const float*)d_in[15];  // [64,32]
  const float* fc2_b    = (const float*)d_in[16];  // [32]
  float* out = (float*)d_out;                      // [2048,32]

  // ---- workspace layout ----
  float* ws    = (float*)d_ws;
  float* agg   = ws;                                  // 100000*128 f32
  float* h_pre = agg   + (size_t)NNODES * INF;        // 100000*64 f32
  float* h_bn  = h_pre + (size_t)NNODES * HID;        // 100000*64 f32
  float* mean  = h_bn  + (size_t)NNODES * HID;        // 64
  float* rstd  = mean + HID;                          // 64
  float* g     = rstd + HID;                          // 2048*64 f32
  unsigned short* us    = (unsigned short*)(g + (size_t)NGRAPH * HID);
  unsigned short* aggb  = us;                               // 100000*128 bf16
  unsigned short* h_midb= aggb   + (size_t)NNODES * INF;    // 100000*64 bf16
  unsigned short* gb    = h_midb + (size_t)NNODES * HID;    // 2048*64 bf16
  unsigned short* g1b   = gb     + (size_t)NGRAPH * HID;    // 2048*64 bf16
  unsigned short* wt    = g1b    + (size_t)NGRAPH * HID;    // weights, bf16 [N][K]
  unsigned short* wt_c1w1 = wt;                 // 64*128
  unsigned short* wt_c1w2 = wt_c1w1 + 64 * 128; // 64*64
  unsigned short* wt_cw1  = wt_c1w2 + 64 * 64;  // 4*64*64
  unsigned short* wt_cw2  = wt_cw1 + 4 * 64 * 64;
  unsigned short* wt_fc1  = wt_cw2 + 4 * 64 * 64;
  unsigned short* wt_fc2  = wt_fc1 + 64 * 64;   // 32*64

  // ---- one-time weight convert + transpose (tiny) ----
  wconv_t<<<32, 256, 0, stream>>>(c1w1, wt_c1w1, INF, HID);
  wconv_t<<<16, 256, 0, stream>>>(c1w2, wt_c1w2, HID, HID);
  for (int i = 0; i < 4; ++i) {
    wconv_t<<<16, 256, 0, stream>>>(cw1 + (size_t)i * HID * HID,
                                    wt_cw1 + (size_t)i * HID * HID, HID, HID);
    wconv_t<<<16, 256, 0, stream>>>(cw2 + (size_t)i * HID * HID,
                                    wt_cw2 + (size_t)i * HID * HID, HID, HID);
  }
  wconv_t<<<16, 256, 0, stream>>>(fc1_w, wt_fc1, HID, HID);
  wconv_t<<<8, 256, 0, stream>>>(fc2_w, wt_fc2, HID, OUTF);

  const float* h_cur = x;
  for (int layer = 0; layer < 5; ++layer) {
    const int F = (layer == 0) ? INF : HID;
    const unsigned short* w1t = (layer == 0) ? wt_c1w1
                                             : wt_cw1 + (size_t)(layer - 1) * HID * HID;
    const unsigned short* w2t = (layer == 0) ? wt_c1w2
                                             : wt_cw2 + (size_t)(layer - 1) * HID * HID;
    const float* b1 = (layer == 0) ? c1b1 : cb1 + (size_t)(layer - 1) * HID;
    const float* b2 = (layer == 0) ? c1b2 : cb2 + (size_t)(layer - 1) * HID;

    // agg = h ; agg[dst] += h[src] over edges
    size_t n4 = (size_t)NNODES * F / 4;
    copy_f4<<<(int)((n4 + 255) / 256), 256, 0, stream>>>(
        (const float4*)h_cur, (float4*)agg, n4);
    scatter_add_edges<<<8192, 256, 0, stream>>>(h_cur, eidx, agg, NEDGES, F);

    // one conversion pass, then pure-WMMA GEMMs (GEMM1 emits bf16 directly)
    cvt(agg, aggb, (size_t)NNODES * F, stream);
    if (F == INF)
      gemm_wmma<INF><<<gemm_blocks(NNODES, HID), 256, 0, stream>>>(
          aggb, w1t, b1, nullptr, h_midb, NNODES, HID, 1);
    else
      gemm_wmma<HID><<<gemm_blocks(NNODES, HID), 256, 0, stream>>>(
          aggb, w1t, b1, nullptr, h_midb, NNODES, HID, 1);
    gemm_wmma<HID><<<gemm_blocks(NNODES, HID), 256, 0, stream>>>(
        h_midb, w2t, b2, h_pre, nullptr, NNODES, HID, 1);

    // BatchNorm (training-mode batch statistics)
    bn_stats<<<HID, 256, 0, stream>>>(h_pre, NNODES, HID, mean, rstd);
    bn_apply<<<25000, 256, 0, stream>>>(h_pre, h_bn, mean, rstd,
                                        bn_gamma + (size_t)layer * HID,
                                        bn_beta  + (size_t)layer * HID,
                                        (size_t)NNODES * HID, HID);
    h_cur = h_bn;
  }

  // global_add_pool -> [2048,64]
  size_t g4 = (size_t)NGRAPH * HID / 4;
  zero_f4<<<(int)((g4 + 255) / 256), 256, 0, stream>>>((float4*)g, g4);
  pool_add<<<4096, 256, 0, stream>>>(h_cur, batch, g, NNODES, HID);

  // fc1 (+ReLU, bf16 out) -> fc2 (f32 out)
  cvt(g, gb, (size_t)NGRAPH * HID, stream);
  gemm_wmma<HID><<<gemm_blocks(NGRAPH, HID), 256, 0, stream>>>(
      gb, wt_fc1, fc1_b, nullptr, g1b, NGRAPH, HID, 1);
  gemm_wmma<HID><<<gemm_blocks(NGRAPH, OUTF), 256, 0, stream>>>(
      g1b, wt_fc2, fc2_b, out, nullptr, NGRAPH, OUTF, 0);
}
